// BiLSTMTagger_24567212934025
// MI455X (gfx1250) — matrix operand order repaired and checked by reading, not compiled, not run on hardware
//
#include <hip/hip_runtime.h>
#include <hip/hip_bf16.h>

// ---------------------------------------------------------------------------
// BiLSTM tagger for MI455X (gfx1250, wave32, WMMA + Tensor Data Mover).
// bf16 operands + f32 accumulation via v_wmma_f32_16x16x32_bf16.
// Big GEMMs stage operand tiles to LDS with TENSOR_LOAD_TO_LDS (TDM),
// double-buffered, with DMA-inserted LDS row padding for bank-conflict-free
// ds reads. The serial LSTM step kernel keeps direct loads (W_hh lives in L2).
// ---------------------------------------------------------------------------

typedef __attribute__((ext_vector_type(16))) __bf16 v16bf;
typedef __attribute__((ext_vector_type(8)))  __bf16 v8bf;
typedef __attribute__((ext_vector_type(8)))  float  v8f;
typedef __attribute__((ext_vector_type(4)))  unsigned int v4u;
typedef __attribute__((ext_vector_type(8)))  int v8i;
typedef __attribute__((ext_vector_type(4)))  int v4i;

#define B_   64
#define T_   256
#define E_   256
#define H_   512
#define G4H  2048      // 4*H
#define O_   50
#define OPAD 64
#define LDS_STRIDE 40  // 32 bf16 payload + 8 bf16 (16B) DMA pad per row

// ----------------------------- WMMA helpers -------------------------------
// A fragment: 16x32 (MxK) tile of row-major A (ld = K).
// lanes 0-15 -> rows 0-15 with K {0..7,16..23}; lanes 16-31 same rows with
// K {8..15,24..31}.
__device__ __forceinline__ v16bf load_a_frag(const __bf16* A, int ld, int row0,
                                             int k0, int lane) {
  int r  = lane & 15;
  int kh = (lane >> 4) << 3;               // 0 or 8
  const __bf16* p = A + (size_t)(row0 + r) * ld + (k0 + kh);
  union { v16bf v; v8bf h[2]; } u;
  u.h[0] = *(const v8bf*)p;                // K = kh .. kh+7
  u.h[1] = *(const v8bf*)(p + 16);         // K = kh+16 .. kh+23
  return u.v;
}

// B fragment: 32x16 (KxN) taken from W stored (N x K) row-major (ld = K):
// lane n (0-15) holds column n, K {0..7,16..23}; lane n+16 holds column n, K {8..15,24..31}.
__device__ __forceinline__ v16bf load_b_frag(const __bf16* W, int ld, int n0,
                                             int k0, int lane) {
  int c  = lane & 15;
  int kh = (lane >> 4) << 3;               // 0 or 8
  const __bf16* p = W + (size_t)(n0 + c) * ld + (k0 + kh);
  union { v16bf v; v8bf h[2]; } u;
  u.h[0] = *(const v8bf*)p;
  u.h[1] = *(const v8bf*)(p + 16);
  return u.v;
}

// Same fragments out of an LDS-staged tile with padded row stride.
__device__ __forceinline__ v16bf lds_a_frag(const __bf16* S, int row0, int lane) {
  int r  = lane & 15;
  int kh = (lane >> 4) << 3;
  const __bf16* p = S + (row0 + r) * LDS_STRIDE + kh;
  union { v16bf v; v8bf h[2]; } u;
  u.h[0] = *(const v8bf*)p;
  u.h[1] = *(const v8bf*)(p + 16);
  return u.v;
}
__device__ __forceinline__ v16bf lds_b_frag(const __bf16* S, int n0, int lane) {
  int c  = lane & 15;
  int kh = (lane >> 4) << 3;
  const __bf16* p = S + (n0 + c) * LDS_STRIDE + kh;
  union { v16bf v; v8bf h[2]; } u;
  u.h[0] = *(const v8bf*)p;
  u.h[1] = *(const v8bf*)(p + 16);
  return u.v;
}

__device__ __forceinline__ v8f wmma_bf16(v16bf a, v16bf b, v8f c) {
  return __builtin_amdgcn_wmma_f32_16x16x32_bf16(
      /*neg_a=*/false, a, /*neg_b=*/false, b,
      /*c_mod=*/(short)0, c, /*reuse_a=*/false, /*reuse_b=*/false);
}

__device__ __forceinline__ float sigm(float x) {
  return 1.0f / (1.0f + __expf(-x));
}
__device__ __forceinline__ float tanh_fast(float x) {
  float e = __expf(2.0f * x);
  return 1.0f - 2.0f / (e + 1.0f);       // saturates to +-1
}

// ----------------------- TDM 2D tile -> LDS issue --------------------------
// D# per CDNA5 ISA ch.8: group0 {count=1, lds_addr, global_addr, type=2},
// group1 {data_size=2B, pad 16dw->+4dw, tensor dims, tile dims, dim0 stride}.
// 2D tile: groups 2/3 (and trailing group) zero. Issued by one wave (TDM
// ignores EXEC), tracked by TENSORcnt. 6-arg builtin form (clang-23 lane).
__device__ __forceinline__ void tdm_load_2d(unsigned lds_addr,
                                            const __bf16* gaddr,
                                            unsigned tensor_d0,
                                            unsigned tensor_d1,
                                            unsigned tile_d0,
                                            unsigned tile_d1,
                                            unsigned long long stride_elems) {
  unsigned long long ga = (unsigned long long)(uintptr_t)gaddr;
  v4u g0;
  g0[0] = 1u;                                               // count=1
  g0[1] = lds_addr;                                         // lds byte addr
  g0[2] = (unsigned)ga;                                     // gaddr[31:0]
  g0[3] = (unsigned)((ga >> 32) & 0x01ffffffu) | (2u << 30);// gaddr[56:32]|type=2
  v8i g1;
  g1[0] = (int)((1u << 16)      // data_size = 2 bytes
              | (1u << 20)      // pad_enable
              | (3u << 22)      // pad_interval: 16 dwords (64B = one row)
              | (3u << 25));    // pad_amount:   4 dwords (16B)
  g1[1] = (int)((tensor_d0 & 0xffffu) << 16);
  g1[2] = (int)(((tensor_d0 >> 16) & 0xffffu) | ((tensor_d1 & 0xffffu) << 16));
  g1[3] = (int)(((tensor_d1 >> 16) & 0xffffu) | ((tile_d0 & 0xffffu) << 16));
  g1[4] = (int)(tile_d1 & 0xffffu);                         // tile_dim2 = 0
  g1[5] = (int)(unsigned)(stride_elems & 0xffffffffu);
  g1[6] = (int)(unsigned)((stride_elems >> 32) & 0xffffu);  // dim1 stride = 0
  g1[7] = 0;
  v4i z4 = {0, 0, 0, 0};
  v8i z8 = {0, 0, 0, 0, 0, 0, 0, 0};
  __builtin_amdgcn_tensor_load_to_lds(g0, g1, z4, z4, z8, 0);
}

// ------------------- GEMM + bias, TDM-staged operands ----------------------
// C[M,N] (f32) = A[M,K] (bf16) * W[N,K]^T (bf16) + bias[N]
// block: 256 thr = 8 waves; tile = 128(M) x 64(N); wave -> 16x64 strip.
// Per 32-wide k-step the TDM leader DMAs a 128x32 A tile and a 64x32 W tile
// into LDS (double buffered, padded rows), everyone computes from LDS.
__global__ __launch_bounds__(256) void gemm_bias_kernel(
    const __bf16* __restrict__ A, const __bf16* __restrict__ W,
    const float* __restrict__ bias, float* __restrict__ C,
    int M, int N, int K) {
  __shared__ __align__(128) __bf16 sA[2][128 * LDS_STRIDE];
  __shared__ __align__(128) __bf16 sB[2][64 * LDS_STRIDE];
  int tid = threadIdx.x, w = tid >> 5, lane = tid & 31;
  int m0 = blockIdx.y * 128;
  int n0 = blockIdx.x * 64;
  bool leader = (__builtin_amdgcn_readfirstlane(w) == 0);

  const __bf16* Abase = A + (size_t)m0 * K;
  const __bf16* Wbase = W + (size_t)n0 * K;
  if (leader) {
    tdm_load_2d((unsigned)(uintptr_t)&sA[0][0], Abase, (unsigned)K,
                (unsigned)M, 32u, 128u, (unsigned long long)K);
    tdm_load_2d((unsigned)(uintptr_t)&sB[0][0], Wbase, (unsigned)K,
                (unsigned)N, 32u, 64u, (unsigned long long)K);
  }

  v8f acc[4] = {};
  int nsteps = K >> 5;
  for (int i = 0; i < nsteps; ++i) {
    int cur = i & 1, nxt = cur ^ 1;
    if (leader) __builtin_amdgcn_s_wait_tensorcnt(0);
    __syncthreads();                       // buf[cur] ready; buf[nxt] free
    if (leader && (i + 1 < nsteps)) {
      int k0 = (i + 1) << 5;
      tdm_load_2d((unsigned)(uintptr_t)&sA[nxt][0], Abase + k0, (unsigned)K,
                  (unsigned)M, 32u, 128u, (unsigned long long)K);
      tdm_load_2d((unsigned)(uintptr_t)&sB[nxt][0], Wbase + k0, (unsigned)K,
                  (unsigned)N, 32u, 64u, (unsigned long long)K);
    }
    v16bf a = lds_a_frag(sA[cur], w * 16, lane);
    #pragma unroll
    for (int j = 0; j < 4; ++j) {
      v16bf b = lds_b_frag(sB[cur], j * 16, lane);
      acc[j] = wmma_bf16(a, b, acc[j]);
    }
  }

  int col  = lane & 15;
  int rsel = (lane >> 4) * 8;              // VGPR v holds row v + rsel
  #pragma unroll
  for (int j = 0; j < 4; ++j) {
    int n = n0 + j * 16 + col;
    float bv = bias[n];
    #pragma unroll
    for (int v = 0; v < 8; ++v)
      C[(size_t)(m0 + w * 16 + rsel + v) * N + n] = acc[j][v] + bv;
  }
}

// --------------------- fused LSTM recurrent step ---------------------------
// One launch per timestep. Block j0 owns hidden slice [j0, j0+64) and computes
// all four gate segments for it: 64(batch) x 256 of h_in @ W_hh^T, adds the
// precomputed xg, applies activations, updates c, writes h (bf16 ping-pong)
// and the (forward, batch-flipped backward) concat output.
__global__ __launch_bounds__(256) void lstm_step_kernel(
    const float* __restrict__ xg, const __bf16* __restrict__ Whh,
    const __bf16* __restrict__ h_in, __bf16* __restrict__ h_out,
    float* __restrict__ c, __bf16* __restrict__ outcat, int t) {
  __shared__ float gl[B_ * 256];           // [batch][gate*64 + j] pre-acts
  int tid = threadIdx.x, w = tid >> 5, lane = tid & 31;
  int j0 = blockIdx.x * 64;
  int mt = w & 3, ngrp = w >> 2;           // wave -> 16 rows x 8 N-tiles
  int m0 = mt * 16;

  v8f acc[8] = {};
  for (int k0 = 0; k0 < H_; k0 += 32) {
    v16bf a = load_a_frag(h_in, H_, m0, k0, lane);
    #pragma unroll
    for (int nt = 0; nt < 8; ++nt) {
      int idx = ngrp * 8 + nt;                       // 0..15 over 4 gates x 4
      int n_global = (idx >> 2) * H_ + j0 + (idx & 3) * 16;
      v16bf b = load_b_frag(Whh, H_, n_global, k0, lane);
      acc[nt] = wmma_bf16(a, b, acc[nt]);
    }
  }

  int col = lane & 15, rsel = (lane >> 4) * 8;
  #pragma unroll
  for (int nt = 0; nt < 8; ++nt) {
    int idx  = ngrp * 8 + nt;
    int gate = idx >> 2, sub = idx & 3;
    int colj = sub * 16 + col;
    int nglob = gate * H_ + j0 + colj;
    #pragma unroll
    for (int v = 0; v < 8; ++v) {
      int b = m0 + rsel + v;
      float val = acc[nt][v] + xg[((size_t)b * T_ + t) * G4H + nglob];
      gl[b * 256 + gate * 64 + colj] = val;
    }
  }
  __syncthreads();

  for (int e = tid; e < B_ * 64; e += 256) {
    int b = e >> 6, j = e & 63;
    const float* g = &gl[b * 256];
    float ig = sigm(g[j]);
    float fg = sigm(g[64 + j]);
    float gg = tanh_fast(g[128 + j]);
    float og = sigm(g[192 + j]);
    int hj = j0 + j;
    float cv = fg * c[b * H_ + hj] + ig * gg;
    c[b * H_ + hj] = cv;
    float hv = og * tanh_fast(cv);
    __bf16 hb = (__bf16)hv;
    h_out[b * H_ + hj] = hb;
    outcat[((size_t)b * T_ + t) * (2 * H_) + hj] = hb;                 // fwd
    outcat[((size_t)(B_ - 1 - b) * T_ + t) * (2 * H_) + H_ + hj] = hb; // flip
  }
}

// ----------------------------- small kernels -------------------------------
__global__ void embed_kernel(const int* __restrict__ x,
                             const float* __restrict__ emb,
                             __bf16* __restrict__ out) {
  int i = blockIdx.x * 256 + threadIdx.x;  // B*T*E
  int bt = i >> 8, e = i & 255;
  int id = x[bt];
  out[i] = (__bf16)emb[(size_t)id * E_ + e];
}

__global__ void f32_to_bf16_kernel(const float* __restrict__ in,
                                   __bf16* __restrict__ out, int n) {
  int i = blockIdx.x * 256 + threadIdx.x;
  if (i < n) out[i] = (__bf16)in[i];
}

__global__ void copy_f32_kernel(const float* __restrict__ in,
                                float* __restrict__ out, int n) {
  int i = blockIdx.x * 256 + threadIdx.x;
  if (i < n) out[i] = in[i];
}

__global__ void pad_fc_kernel(const float* __restrict__ Wfc,
                              const float* __restrict__ bfc,
                              __bf16* __restrict__ Wp,
                              float* __restrict__ bp) {
  int i = blockIdx.x * 256 + threadIdx.x;  // OPAD*1024
  int r = i >> 10, cix = i & 1023;
  Wp[i] = (r < O_) ? (__bf16)Wfc[r * 1024 + cix] : (__bf16)0.0f;
  if (i < OPAD) bp[i] = (i < O_) ? bfc[i] : 0.0f;
}

// one wave per row; 50 valid of 64 padded columns
__global__ __launch_bounds__(256) void logsoftmax_kernel(
    const float* __restrict__ logits, float* __restrict__ out) {
  int tid = threadIdx.x, w = tid >> 5, lane = tid & 31;
  int row = blockIdx.x * 8 + w;
  const float* p = logits + (size_t)row * OPAD;
  float v0 = (lane < O_) ? p[lane] : -1e30f;
  float v1 = (lane + 32 < O_) ? p[lane + 32] : -1e30f;
  float m = fmaxf(v0, v1);
  for (int s = 16; s > 0; s >>= 1) m = fmaxf(m, __shfl_xor(m, s, 32));
  float s0 = (lane < O_) ? __expf(v0 - m) : 0.0f;
  float s1 = (lane + 32 < O_) ? __expf(v1 - m) : 0.0f;
  float s = s0 + s1;
  for (int sh = 16; sh > 0; sh >>= 1) s += __shfl_xor(s, sh, 32);
  float lse = m + __logf(s);
  if (lane < O_)      out[(size_t)row * O_ + lane]      = v0 - lse;
  if (lane + 32 < O_) out[(size_t)row * O_ + lane + 32] = v1 - lse;
}

// ------------------------------- launcher ----------------------------------
extern "C" void kernel_launch(void* const* d_in, const int* in_sizes, int n_in,
                              void* d_out, int out_size, void* d_ws,
                              size_t ws_size, hipStream_t stream) {
  (void)in_sizes; (void)n_in; (void)out_size; (void)ws_size;
  const int*   x    = (const int*)  d_in[0];
  const float* emb  = (const float*)d_in[1];
  const float* Wih1 = (const float*)d_in[2];
  const float* Whh1 = (const float*)d_in[3];
  const float* b1   = (const float*)d_in[4];
  const float* h01  = (const float*)d_in[5];
  const float* c01  = (const float*)d_in[6];
  const float* Wih2 = (const float*)d_in[7];
  const float* Whh2 = (const float*)d_in[8];
  const float* b2   = (const float*)d_in[9];
  const float* h02  = (const float*)d_in[10];
  const float* c02  = (const float*)d_in[11];
  const float* Wfc  = (const float*)d_in[12];
  const float* bfc  = (const float*)d_in[13];
  float* out = (float*)d_out;

  // workspace carve-up (~224 MB total), 256B aligned
  char* ws = (char*)d_ws;
  size_t off = 0;
  auto alloc = [&](size_t bytes) -> char* {
    char* p = ws + off;
    off += (bytes + 255) & ~(size_t)255;
    return p;
  };
  __bf16* Wih1b = (__bf16*)alloc((size_t)G4H * E_ * 2);
  __bf16* Whh1b = (__bf16*)alloc((size_t)G4H * H_ * 2);
  __bf16* Wih2b = (__bf16*)alloc((size_t)G4H * 2 * H_ * 2);
  __bf16* Whh2b = (__bf16*)alloc((size_t)G4H * H_ * 2);
  __bf16* Wfcp  = (__bf16*)alloc((size_t)OPAD * 1024 * 2);
  float*  bfcp  = (float*) alloc(OPAD * 4);
  __bf16* xemb  = (__bf16*)alloc((size_t)B_ * T_ * E_ * 2);
  float*  xg    = (float*) alloc((size_t)B_ * T_ * G4H * 4);   // reused L1+L2
  __bf16* cat1  = (__bf16*)alloc((size_t)B_ * T_ * 2 * H_ * 2);
  __bf16* cat2  = (__bf16*)alloc((size_t)B_ * T_ * 2 * H_ * 2);
  __bf16* hping = (__bf16*)alloc((size_t)B_ * H_ * 2);
  __bf16* hpong = (__bf16*)alloc((size_t)B_ * H_ * 2);
  float*  cbuf  = (float*) alloc((size_t)B_ * H_ * 4);
  float*  logitsp = (float*)alloc((size_t)B_ * T_ * OPAD * 4);

  auto cvt = [&](const float* src, __bf16* dst, int n) {
    f32_to_bf16_kernel<<<(n + 255) / 256, 256, 0, stream>>>(src, dst, n);
  };

  // weight conversion + embedding
  cvt(Wih1, Wih1b, G4H * E_);
  cvt(Whh1, Whh1b, G4H * H_);
  cvt(Wih2, Wih2b, G4H * 2 * H_);
  cvt(Whh2, Whh2b, G4H * H_);
  pad_fc_kernel<<<(OPAD * 1024) / 256, 256, 0, stream>>>(Wfc, bfc, Wfcp, bfcp);
  embed_kernel<<<(B_ * T_ * E_) / 256, 256, 0, stream>>>(x, emb, xemb);

  // ---- layer 1 ----
  gemm_bias_kernel<<<dim3(G4H / 64, (B_ * T_) / 128), 256, 0, stream>>>(
      xemb, Wih1b, b1, xg, B_ * T_, G4H, E_);
  cvt(h01, hping, B_ * H_);
  copy_f32_kernel<<<(B_ * H_) / 256, 256, 0, stream>>>(c01, cbuf, B_ * H_);
  for (int t = 0; t < T_; ++t) {
    const __bf16* hin = (t & 1) ? hpong : hping;
    __bf16* hout      = (t & 1) ? hping : hpong;
    lstm_step_kernel<<<H_ / 64, 256, 0, stream>>>(xg, Whh1b, hin, hout, cbuf,
                                                  cat1, t);
  }

  // ---- layer 2 ----
  gemm_bias_kernel<<<dim3(G4H / 64, (B_ * T_) / 128), 256, 0, stream>>>(
      cat1, Wih2b, b2, xg, B_ * T_, G4H, 2 * H_);
  cvt(h02, hping, B_ * H_);
  copy_f32_kernel<<<(B_ * H_) / 256, 256, 0, stream>>>(c02, cbuf, B_ * H_);
  for (int t = 0; t < T_; ++t) {
    const __bf16* hin = (t & 1) ? hpong : hping;
    __bf16* hout      = (t & 1) ? hping : hpong;
    lstm_step_kernel<<<H_ / 64, 256, 0, stream>>>(xg, Whh2b, hin, hout, cbuf,
                                                  cat2, t);
  }

  // ---- FC (padded to 64 cols) + log_softmax ----
  gemm_bias_kernel<<<dim3(OPAD / 64, (B_ * T_) / 128), 256, 0, stream>>>(
      cat2, Wfcp, bfcp, logitsp, B_ * T_, OPAD, 2 * H_);
  logsoftmax_kernel<<<(B_ * T_) / 8, 256, 0, stream>>>(logitsp, out);
}
